// ScaledDotProductAttention_15410342658149
// MI455X (gfx1250) — matrix operand order repaired and checked by reading, not compiled, not run on hardware
//
#include <hip/hip_runtime.h>

// Problem constants (from reference): B=2, H=16, S=2048, D=64, temperature=32
#define S_LEN   2048
#define DHEAD   64
#define NBH     32          // B*H
#define QTILE   64          // q rows per workgroup
#define NWAVES  4           // 128 threads, wave32
#define SCALE   (1.0f/32.0f)

typedef __attribute__((ext_vector_type(16))) _Float16 v16h;
typedef __attribute__((ext_vector_type(8)))  float    v8f;
typedef __attribute__((ext_vector_type(4)))  float    f32x4;  // clang vector (NT-store ok)
typedef __attribute__((ext_vector_type(4)))  _Float16 h4;
typedef __attribute__((ext_vector_type(4)))  unsigned u32x4;

// Trivial union: both members are ext-vector types (no ctors).
union AFrag { v16h h; u32x4 u[2]; };

// One-shot 16-lane reductions (stay within each wave32 half; xor of bits 0..3)
__device__ __forceinline__ float red_max16(float v) {
  v = fmaxf(v, __shfl_xor(v, 1, 32));
  v = fmaxf(v, __shfl_xor(v, 2, 32));
  v = fmaxf(v, __shfl_xor(v, 4, 32));
  v = fmaxf(v, __shfl_xor(v, 8, 32));
  return v;
}
__device__ __forceinline__ float red_sum16(float v) {
  v += __shfl_xor(v, 1, 32);
  v += __shfl_xor(v, 2, 32);
  v += __shfl_xor(v, 4, 32);
  v += __shfl_xor(v, 8, 32);
  return v;
}

__global__ __launch_bounds__(128)
void sdpa_wmma_kernel(const float* __restrict__ q,
                      const float* __restrict__ k,
                      const float* __restrict__ v,
                      const int*   __restrict__ mask,
                      float* __restrict__ outO,
                      float* __restrict__ attn)
{
  __shared__ __align__(16) _Float16 smemK[64 * 64];          // Q/K staging tile (8KB)
  __shared__ __align__(16) _Float16 smemV[64 * 64];          // V transposed [d][key] (8KB)
  __shared__ __align__(16) _Float16 smemP[NWAVES * 16 * 64]; // per-wave P tile (8KB)

  const int tid  = threadIdx.x;
  const int lane = tid & 31;
  const int wave = tid >> 5;
  const int ln   = lane & 15;         // column / row-within-16
  const int hb   = (lane >> 4) & 1;   // lane-half selector

  const int bid = blockIdx.x;
  const int bh  = bid >> 5;           // (b*H + h) in 0..31
  const int qt  = bid & 31;           // q tile index
  const int bb  = bh >> 4;            // batch index

  const size_t qkvBase = (size_t)bh * S_LEN * DHEAD;
  const float* qg   = q + qkvBase + (size_t)qt * QTILE * DHEAD;
  const int*   mrow = mask + (size_t)bb * S_LEN;
  float* attnW = attn + (size_t)bh * S_LEN * S_LEN
                      + (size_t)(qt * QTILE + wave * 16) * S_LEN;
  float* outW  = outO + qkvBase + (size_t)(qt * QTILE + wave * 16) * DHEAD;

  // ---------------- stage Q tile to LDS as f16, folded 1/temperature ----------------
  {
    const float4* q4 = (const float4*)qg;          // 64x64 f32 = 1024 float4
    #pragma unroll
    for (int i = 0; i < 8; ++i) {
      int idx = tid + i * 128;
      float4 f = q4[idx];
      h4 hh = { (_Float16)(f.x * SCALE), (_Float16)(f.y * SCALE),
                (_Float16)(f.z * SCALE), (_Float16)(f.w * SCALE) };
      *(h4*)(smemK + (idx >> 4) * 64 + (idx & 15) * 4) = hh;
    }
  }
  __syncthreads();

  // This wave's Q A-fragments (16 rows x 64 depth => two 16x32 frags).
  AFrag A0, A1;
  {
    const _Float16* base = smemK + (wave * 16 + ln) * 64 + hb * 8;
    A0.u[0] = *(const u32x4*)(base + 0);
    A0.u[1] = *(const u32x4*)(base + 16);
    A1.u[0] = *(const u32x4*)(base + 32);
    A1.u[1] = *(const u32x4*)(base + 48);
  }
  __syncthreads();

  // ---------------- sweep A: QK^T -> raw scores + per-lane partial max --------------
  float mp[8];
  #pragma unroll
  for (int r = 0; r < 8; ++r) mp[r] = -3.0e38f;

  for (int kt = 0; kt < S_LEN / 64; ++kt) {
    const float* kbase = k + qkvBase + (size_t)kt * 64 * DHEAD;
    if (kt + 1 < S_LEN / 64)
      __builtin_prefetch(kbase + 64 * DHEAD + tid * 32, 0, 0);

    const float4* k4 = (const float4*)kbase;
    #pragma unroll
    for (int i = 0; i < 8; ++i) {
      int idx = tid + i * 128;
      float4 f = k4[idx];
      h4 hh = { (_Float16)f.x, (_Float16)f.y, (_Float16)f.z, (_Float16)f.w };
      *(h4*)(smemK + (idx >> 4) * 64 + (idx & 15) * 4) = hh;
    }
    __syncthreads();

    // Preload ALL 8 B-fragments into distinct live values -> staggered dscnt waits.
    AFrag B[8];
    #pragma unroll
    for (int nsub = 0; nsub < 4; ++nsub) {
      const _Float16* kb = smemK + (nsub * 16 + ln) * 64 + hb * 8;
      B[2*nsub + 0].u[0] = *(const u32x4*)(kb + 0);
      B[2*nsub + 0].u[1] = *(const u32x4*)(kb + 16);
      B[2*nsub + 1].u[0] = *(const u32x4*)(kb + 32);
      B[2*nsub + 1].u[1] = *(const u32x4*)(kb + 48);
    }

    v8f c[4];
    #pragma unroll
    for (int nsub = 0; nsub < 4; ++nsub) {
      v8f t = {};
      t = __builtin_amdgcn_wmma_f32_16x16x32_f16(false, A0.h, false, B[2*nsub+0].h,
                                                 (short)0, t, false, false);
      t = __builtin_amdgcn_wmma_f32_16x16x32_f16(false, A1.h, false, B[2*nsub+1].h,
                                                 (short)0, t, false, false);
      c[nsub] = t;
    }

    #pragma unroll
    for (int nsub = 0; nsub < 4; ++nsub) {
      const int key = kt * 64 + nsub * 16 + ln;
      const int mv  = mrow[key];
      #pragma unroll
      for (int r = 0; r < 8; ++r) {
        float s = (mv == 0) ? -1.0e9f : c[nsub][r];
        attnW[(size_t)(r + hb * 8) * S_LEN + key] = s;  // raw score scratch (L2-hot)
        mp[r] = fmaxf(mp[r], s);                        // per-lane only, no shuffles
      }
    }
    __syncthreads();
  }

  // single cross-lane max reduction per row for the whole kernel
  float mrun[8];
  #pragma unroll
  for (int r = 0; r < 8; ++r) mrun[r] = red_max16(mp[r]);

  // ---------------- sweep B: s -> e = exp(s - m), per-lane partial sums -------------
  float lp[8];
  #pragma unroll
  for (int r = 0; r < 8; ++r) lp[r] = 0.0f;

  for (int c4i = 0; c4i < S_LEN / 64; ++c4i) {      // each lane: 4 consecutive cols
    const int colbase = c4i * 64 + ln * 4;
    #pragma unroll
    for (int r = 0; r < 8; ++r) {
      float* rowp = attnW + (size_t)(r + hb * 8) * S_LEN + colbase;
      f32x4 s4 = *(const f32x4*)rowp;
      float e0 = __expf(s4.x - mrun[r]);
      float e1 = __expf(s4.y - mrun[r]);
      float e2 = __expf(s4.z - mrun[r]);
      float e3 = __expf(s4.w - mrun[r]);
      lp[r] += (e0 + e1) + (e2 + e3);
      f32x4 e4 = { e0, e1, e2, e3 };
      *(f32x4*)rowp = e4;                           // overwrite with exp values
    }
  }

  // single cross-lane sum reduction per row, then reciprocal
  float rl[8];
  #pragma unroll
  for (int r = 0; r < 8; ++r) rl[r] = 1.0f / red_sum16(lp[r]);

  // ---------------- sweep C: p = e/l (NT float4 store) + P*V via WMMA ---------------
  v8f acc[4] = {};
  _Float16* smemPw = smemP + wave * 16 * 64;

  for (int kt = 0; kt < S_LEN / 64; ++kt) {
    const float* vbase = v + qkvBase + (size_t)kt * 64 * DHEAD;
    if (kt + 1 < S_LEN / 64)
      __builtin_prefetch(vbase + 64 * DHEAD + tid * 32, 0, 0);

    // stage V key-tile (64x64) transposed into [d][key] f16
    const float4* v4 = (const float4*)vbase;
    #pragma unroll
    for (int i = 0; i < 8; ++i) {
      int idx = tid + i * 128;        // 1024 float4 total
      float4 f = v4[idx];
      int row = idx >> 4;             // key 0..63
      int cc  = (idx & 15) * 4;       // d
      smemV[(cc + 0) * 64 + row] = (_Float16)f.x;
      smemV[(cc + 1) * 64 + row] = (_Float16)f.y;
      smemV[(cc + 2) * 64 + row] = (_Float16)f.z;
      smemV[(cc + 3) * 64 + row] = (_Float16)f.w;
    }
    __syncthreads();

    // normalize 64 columns: float4 per (lane,row); write p (NT) + f16 P tile (b64)
    const int colbase = kt * 64 + ln * 4;
    #pragma unroll
    for (int r = 0; r < 8; ++r) {
      const int M = r + hb * 8;
      float* rowp = attnW + (size_t)M * S_LEN + colbase;
      f32x4 e4 = *(const f32x4*)rowp;               // same-wave RAW, in-order
      f32x4 p4 = e4 * rl[r];
      __builtin_nontemporal_store(p4, (f32x4*)rowp);    // final attention, evict early
      h4 ph = { (_Float16)p4.x, (_Float16)p4.y, (_Float16)p4.z, (_Float16)p4.w };
      *(h4*)(smemPw + M * 64 + ln * 4) = ph;        // wave-local LDS, DS in-order
    }

    // P A-fragments (16 rows x 64 keys => two 16x32 frags)
    AFrag PA0, PA1;
    {
      const _Float16* pb = smemPw + ln * 64 + hb * 8;
      PA0.u[0] = *(const u32x4*)(pb + 0);
      PA0.u[1] = *(const u32x4*)(pb + 16);
      PA1.u[0] = *(const u32x4*)(pb + 32);
      PA1.u[1] = *(const u32x4*)(pb + 48);
    }

    // V B-fragments from transposed tile; two depth halves per d-subtile
    #pragma unroll
    for (int dsub = 0; dsub < 4; ++dsub) {
      AFrag BV0, BV1;
      const _Float16* vb = smemV + (dsub * 16 + ln) * 64 + hb * 8;
      BV0.u[0] = *(const u32x4*)(vb + 0);
      BV0.u[1] = *(const u32x4*)(vb + 16);
      BV1.u[0] = *(const u32x4*)(vb + 32);
      BV1.u[1] = *(const u32x4*)(vb + 48);
      acc[dsub] = __builtin_amdgcn_wmma_f32_16x16x32_f16(false, PA0.h, false, BV0.h,
                                                         (short)0, acc[dsub],
                                                         false, false);
      acc[dsub] = __builtin_amdgcn_wmma_f32_16x16x32_f16(false, PA1.h, false, BV1.h,
                                                         (short)0, acc[dsub],
                                                         false, false);
    }
    __syncthreads();
  }

  // ---------------- write output [16 rows x 64 d] per wave --------------------------
  #pragma unroll
  for (int dsub = 0; dsub < 4; ++dsub) {
    #pragma unroll
    for (int r = 0; r < 8; ++r) {
      __builtin_nontemporal_store(acc[dsub][r],
          outW + (size_t)(r + hb * 8) * DHEAD + dsub * 16 + ln);
    }
  }
}

extern "C" void kernel_launch(void* const* d_in, const int* in_sizes, int n_in,
                              void* d_out, int out_size, void* d_ws, size_t ws_size,
                              hipStream_t stream) {
  const float* q    = (const float*)d_in[0];
  const float* k    = (const float*)d_in[1];
  const float* v    = (const float*)d_in[2];
  const int*   mask = (const int*)d_in[3];

  float* outO = (float*)d_out;                               // [B,H,S,D]
  float* attn = outO + (size_t)NBH * S_LEN * DHEAD;          // [B,H,S,S]

  dim3 grid(NBH * (S_LEN / QTILE));   // 32 * 32 = 1024 workgroups
  dim3 block(NWAVES * 32);            // 128 threads = 4 wave32

  hipLaunchKernelGGL(sdpa_wmma_kernel, grid, block, 0, stream,
                     q, k, v, mask, outO, attn);
}
